// Graph_Attention_Union_22316650070603
// MI455X (gfx1250) — matrix operand order repaired and checked by reading, not compiled, not run on hardware
//
#include <hip/hip_runtime.h>
#include <hip/hip_bf16.h>

typedef __attribute__((ext_vector_type(16))) __bf16 v16bf;
typedef __attribute__((ext_vector_type(8)))  __bf16 v8bf;
typedef __attribute__((ext_vector_type(8)))  float  v8f;

#define BN_EPS 1e-5f

// ---------------------------------------------------------------------------
// CDNA5 async global->LDS copy (ASYNCcnt-tracked DMA, CDNA5 ISA).
// Per-lane: LDS[vdst] = MEM[saddr + vaddr] (16 bytes). Inline asm because the
// clang builtins for the async path are not probe-confirmed on both toolchains.
// ---------------------------------------------------------------------------
__device__ __forceinline__ void async_g2l_b128(unsigned ldsAddr,
                                               unsigned long long saddr,
                                               int voff) {
  asm volatile("global_load_async_to_lds_b128 %0, %1, %2"
               :
               : "v"(ldsAddr), "v"(voff), "s"(saddr)
               : "memory");
}
__device__ __forceinline__ void wait_async0() {
  asm volatile("s_wait_asynccnt 0x0" ::: "memory");
}

// ---------------------------------------------------------------------------
// pack f32 -> bf16 with column padding (pad cols = 0 so padded K/N regions
// contribute exactly zero in later WMMA GEMMs)
// ---------------------------------------------------------------------------
__global__ __launch_bounds__(256) void pack_bf16_pad(const float* __restrict__ src,
                                                     __bf16* __restrict__ dst,
                                                     int rows, int cols, int ldDst) {
  long long i = (long long)blockIdx.x * 256 + threadIdx.x;
  long long tot = (long long)rows * ldDst;
  if (i >= tot) return;
  int r = (int)(i / ldDst);
  int c = (int)(i % ldDst);
  float v = (c < cols) ? src[(long long)r * cols + c] : 0.0f;
  dst[i] = (__bf16)v;
}

// ---------------------------------------------------------------------------
// bf16 tiled transpose: src [R x Cc] -> dst [Cc x R], batched over blockIdx.z
// ---------------------------------------------------------------------------
__global__ __launch_bounds__(256) void transpose_bf16(const __bf16* __restrict__ src,
                                                      __bf16* __restrict__ dst,
                                                      int R, int Cc) {
  __shared__ __bf16 tile[32][33];
  long long bb = blockIdx.z;
  src += bb * (long long)R * Cc;
  dst += bb * (long long)R * Cc;
  int c0 = blockIdx.x * 32;
  int r0 = blockIdx.y * 32;
  int tx = threadIdx.x & 31;
  int ty = threadIdx.x >> 5;   // 0..7
#pragma unroll
  for (int j = 0; j < 32; j += 8)
    tile[ty + j][tx] = src[(long long)(r0 + ty + j) * Cc + (c0 + tx)];
  __syncthreads();
#pragma unroll
  for (int j = 0; j < 32; j += 8)
    dst[(long long)(c0 + ty + j) * R + (r0 + tx)] = tile[tx][ty + j];
}

// ---------------------------------------------------------------------------
// Row softmax over f32 logits storing the TRANSPOSED bf16 probability matrix:
// T[m][n] = softmax_row_n(L)[m]. Pad rows/cols written as zero (inert K-dim).
//   grid.x = source row n, grid.y = batch
// ---------------------------------------------------------------------------
__global__ __launch_bounds__(256) void softmax_transpose(
    const float* __restrict__ L, long long sL, int ldL, int ncols, int nrows,
    __bf16* __restrict__ T, long long sT, int ldT, int Mt) {
  int b = blockIdx.y;
  int n = blockIdx.x;
  const float* row = L + (long long)b * sL + (long long)n * ldL;
  __bf16* outcol = T + (long long)b * sT + n;
  __shared__ float red[256];
  int t = threadIdx.x;
  bool valid = (n < nrows);

  float mx = -3.4e38f;
  if (valid)
    for (int m = t; m < ncols; m += 256) mx = fmaxf(mx, row[m]);
  red[t] = mx;
  __syncthreads();
  for (int s = 128; s > 0; s >>= 1) {
    if (t < s) red[t] = fmaxf(red[t], red[t + s]);
    __syncthreads();
  }
  mx = red[0];
  __syncthreads();

  float sum = 0.0f;
  if (valid)
    for (int m = t; m < ncols; m += 256) sum += __expf(row[m] - mx);
  red[t] = sum;
  __syncthreads();
  for (int s = 128; s > 0; s >>= 1) {
    if (t < s) red[t] += red[t + s];
    __syncthreads();
  }
  float inv = 1.0f / red[0];

  for (int m = t; m < Mt; m += 256) {
    float p = (valid && m < ncols) ? __expf(row[m] - mx) * inv : 0.0f;
    outcol[(long long)m * ldT] = (__bf16)p;
  }
}

// ---------------------------------------------------------------------------
// WMMA bf16 GEMM:  C[M,N] = A[M,K] * B[K,N]   (row-major)
//   block = 256 threads = 8 waves; block tile 128x64; wave tile 32x32
//   (2x2 of v_wmma_f32_16x16x32_bf16); K stepped by 32.
//   A tile: async global->LDS DMA (double buffered, 1 barrier per k-step).
//   B tile: register transpose into Bst[n][k] so B fragments are contiguous
//           32B runs -> ds_load_b128 (no scalar gather).
// Requires M%128==0, N%64==0, K%32==0 (guaranteed by padded buffers).
// ---------------------------------------------------------------------------
template <int HAS_BIAS, int HAS_BN, int OUT_F32>
__global__ __launch_bounds__(256) void gemm_wmma_bf16(
    const __bf16* __restrict__ Ag, long long sA, int lda,
    const __bf16* __restrict__ Bg, long long sB, int ldb,
    void* __restrict__ Cg, long long sC, int ldc, int Nact, int K,
    const float* __restrict__ bias,
    const float* __restrict__ gamma, const float* __restrict__ beta,
    const float* __restrict__ mean, const float* __restrict__ var) {
  __shared__ __bf16 As[2][128][32];   // row-major A tiles
  __shared__ __bf16 Bst[2][64][48];   // TRANSPOSED B tiles [n][k], 96B rows

  const int bz = blockIdx.z;
  const __bf16* A = Ag + (long long)bz * sA;
  const __bf16* B = Bg + (long long)bz * sB;
  const unsigned long long Abase = (unsigned long long)A;

  const int m0 = blockIdx.y * 128;
  const int n0 = blockIdx.x * 64;
  const int t = threadIdx.x;
  const int lane = t & 31;
  const int wave = t >> 5;
  const int mW = (wave >> 1) * 32;
  const int nW = (wave & 1) * 32;

  // A tile fill: 128x32, 16 bf16/thread as 2 async b128
  const int arow = t >> 1;
  const int acol = (t & 1) * 16;
  // B tile fill: 32x64, 8 bf16/thread (one uint4), stored transposed
  const int brow = t >> 3;
  const int bcol = (t & 7) * 8;

  v8f acc[2][2];
#pragma unroll
  for (int i = 0; i < 2; ++i)
#pragma unroll
    for (int j = 0; j < 2; ++j) {
      v8f z = {0.f, 0.f, 0.f, 0.f, 0.f, 0.f, 0.f, 0.f};
      acc[i][j] = z;
    }

  const int r16 = lane & 15;
  const int kA0 = (lane < 16) ? 0 : 16;  // A frag: lanes 0-15 hold K0..15
  const int kB0 = (lane < 16) ? 0 : 16;  // B frag: lanes 0-15 hold K0..15

  auto issueA = [&](int k0, int buf) {
    unsigned l0 = (unsigned)(size_t)&As[buf][arow][acol];
    int goff = (int)((((long long)(m0 + arow) * lda) + (k0 + acol)) * 2);
    async_g2l_b128(l0, Abase, goff);
    async_g2l_b128(l0 + 16, Abase, goff + 16);
  };
  auto loadB = [&](int k0) -> uint4 {
    return *(const uint4*)(B + (long long)(k0 + brow) * ldb + (n0 + bcol));
  };
  auto storeB = [&](uint4 bv, int buf) {
    union { uint4 u; __bf16 h[8]; } cv;
    cv.u = bv;
#pragma unroll
    for (int j = 0; j < 8; ++j) Bst[buf][bcol + j][brow] = cv.h[j];
  };

  // prologue: stage tile 0
  issueA(0, 0);
  storeB(loadB(0), 0);

  int buf = 0;
  for (int k0 = 0; k0 < K; k0 += 32) {
    wait_async0();       // A tile DMA for 'buf' complete (this wave)
    __syncthreads();     // all waves' A DMA + B ds-stores visible

    if (k0 + 32 < K) {   // prefetch next tile into other buffer
      issueA(k0 + 32, buf ^ 1);
      storeB(loadB(k0 + 32), buf ^ 1);
    }

    v16bf afrag[2], bfrag[2];
#pragma unroll
    for (int i = 0; i < 2; ++i) {
      v8bf lo = *(const v8bf*)&As[buf][mW + 16 * i + r16][kA0];
      v8bf hi = *(const v8bf*)&As[buf][mW + 16 * i + r16][kA0 + 8];
      afrag[i] = __builtin_shufflevector(lo, hi, 0, 1, 2, 3, 4, 5, 6, 7, 8, 9,
                                         10, 11, 12, 13, 14, 15);
    }
#pragma unroll
    for (int j = 0; j < 2; ++j)
      bfrag[j] = *(const v16bf*)&Bst[buf][nW + 16 * j + r16][kB0];

#pragma unroll
    for (int i = 0; i < 2; ++i)
#pragma unroll
      for (int j = 0; j < 2; ++j)
        acc[i][j] = __builtin_amdgcn_wmma_f32_16x16x32_bf16(
            false, afrag[i], false, bfrag[j], (short)0, acc[i][j], false,
            false);
    buf ^= 1;
  }

  // epilogue: C/D layout — VGPR r: row = r (lanes 0-15) / r+8 (16-31), col = lane&15
  const int ccol = lane & 15;
  const int rb = (lane < 16) ? 0 : 8;
  float* Cf = (float*)Cg + (long long)bz * sC;
  __bf16* Cb = (__bf16*)Cg + (long long)bz * sC;
#pragma unroll
  for (int i = 0; i < 2; ++i)
#pragma unroll
    for (int j = 0; j < 2; ++j)
#pragma unroll
      for (int r = 0; r < 8; ++r) {
        int row = m0 + mW + 16 * i + rb + r;
        int col = n0 + nW + 16 * j + ccol;
        float v = acc[i][j][r];
        if (HAS_BIAS) v += bias[row];
        if (HAS_BN) {
          float inv = rsqrtf(var[row] + BN_EPS);
          v = (v - mean[row]) * (gamma[row] * inv) + beta[row];
          v = fmaxf(v, 0.0f);
        }
        if (OUT_F32) {
          if (col < Nact) Cf[(long long)row * ldc + col] = v;
        } else {
          Cb[(long long)row * ldc + col] =
              (col < Nact) ? (__bf16)v : (__bf16)0.0f;
        }
      }
}

// ---------------------------------------------------------------------------
// host-side orchestration
// ---------------------------------------------------------------------------
extern "C" void kernel_launch(void* const* d_in, const int* in_sizes, int n_in,
                              void* d_out, int out_size, void* d_ws,
                              size_t ws_size, hipStream_t stream) {
  (void)in_sizes; (void)n_in; (void)out_size; (void)ws_size;
  const float* zf = (const float*)d_in[0];
  const float* xf = (const float*)d_in[1];
  const float* Wq = (const float*)d_in[2];
  const float* bq = (const float*)d_in[3];
  const float* Wsw = (const float*)d_in[4];
  const float* bs = (const float*)d_in[5];
  const float* Wg = (const float*)d_in[6];
  const float* bg = (const float*)d_in[7];
  const float* g_gamma = (const float*)d_in[8];
  const float* g_beta = (const float*)d_in[9];
  const float* g_mean = (const float*)d_in[10];
  const float* g_var = (const float*)d_in[11];
  const float* Wfi = (const float*)d_in[12];
  const float* bfi = (const float*)d_in[13];
  const float* fi_gamma = (const float*)d_in[14];
  const float* fi_beta = (const float*)d_in[15];
  const float* fi_mean = (const float*)d_in[16];
  const float* fi_var = (const float*)d_in[17];
  float* out = (float*)d_out;

  const int B = 32, C = 256, NX = 961, NZ = 49;
  const int NXP = 1024, NZP = 64, K3 = 768, O = 256;

  size_t off = 0;
  auto carve = [&](size_t bytes) -> void* {
    off = (off + 255) & ~(size_t)255;
    void* p = (char*)d_ws + off;
    off += bytes;
    return p;
  };
  __bf16* xf_bf = (__bf16*)carve((size_t)B * C * NXP * 2);   // [B][C][NXP]
  __bf16* zf_bf = (__bf16*)carve((size_t)B * C * NZP * 2);   // [B][C][NZP]
  __bf16* Wq_bf = (__bf16*)carve((size_t)C * C * 2);
  __bf16* Ws_bf = (__bf16*)carve((size_t)C * C * 2);
  __bf16* Wg_bf = (__bf16*)carve((size_t)C * C * 2);
  __bf16* Wfi_bf = (__bf16*)carve((size_t)O * K3 * 2);
  __bf16* XTcn = (__bf16*)carve((size_t)B * C * NXP * 2);    // xf_trans [C][NXP]
  __bf16* XTnc = (__bf16*)carve((size_t)B * NXP * C * 2);    // xf_trans^T [NXP][C]
  __bf16* ZT = (__bf16*)carve((size_t)B * C * NZP * 2);      // zf_trans [C][NZP]
  __bf16* ZG = (__bf16*)carve((size_t)B * C * NZP * 2);      // zf_g [C][NZP]
  __bf16* F = (__bf16*)carve((size_t)B * K3 * NXP * 2);      // [emb; self_emb; xf_g]
  float* Lself = (float*)carve((size_t)B * NXP * NXP * 4);   // self logits
  float* Lsim = (float*)carve((size_t)B * NXP * NZP * 4);    // cross logits
  __bf16* S_T = (__bf16*)carve((size_t)B * NXP * NXP * 2);   // self probs^T
  __bf16* P_T = (__bf16*)carve((size_t)B * NZP * NXP * 2);   // cross probs^T

  dim3 blk(256);
  auto blocks1d = [](long long n) { return dim3((unsigned)((n + 255) / 256)); };

  // ---- pack inputs/weights to bf16 (zero-padded) ----
  pack_bf16_pad<<<blocks1d((long long)B * C * NXP), blk, 0, stream>>>(
      xf, xf_bf, B * C, NX, NXP);
  pack_bf16_pad<<<blocks1d((long long)B * C * NZP), blk, 0, stream>>>(
      zf, zf_bf, B * C, NZ, NZP);
  pack_bf16_pad<<<blocks1d((long long)C * C), blk, 0, stream>>>(Wq, Wq_bf, C, C, C);
  pack_bf16_pad<<<blocks1d((long long)C * C), blk, 0, stream>>>(Wsw, Ws_bf, C, C, C);
  pack_bf16_pad<<<blocks1d((long long)C * C), blk, 0, stream>>>(Wg, Wg_bf, C, C, C);
  pack_bf16_pad<<<blocks1d((long long)O * K3), blk, 0, stream>>>(Wfi, Wfi_bf, O, K3, K3);

  const long long sXF = (long long)C * NXP, sZF = (long long)C * NZP;
  const long long sF = (long long)K3 * NXP;

  // ---- stage 1: 1x1 convs (GEMMs over channel dim) ----
  gemm_wmma_bf16<1, 0, 0><<<dim3(NXP / 64, C / 128, B), blk, 0, stream>>>(
      Wq_bf, 0, C, xf_bf, sXF, NXP, XTcn, sXF, NXP, NX, C, bq, nullptr,
      nullptr, nullptr, nullptr);
  gemm_wmma_bf16<1, 1, 0><<<dim3(NXP / 64, C / 128, B), blk, 0, stream>>>(
      Wg_bf, 0, C, xf_bf, sXF, NXP, F + (long long)512 * NXP, sF, NXP, NX, C,
      bg, g_gamma, g_beta, g_mean, g_var);
  gemm_wmma_bf16<1, 0, 0><<<dim3(NZP / 64, C / 128, B), blk, 0, stream>>>(
      Ws_bf, 0, C, zf_bf, sZF, NZP, ZT, sZF, NZP, NZ, C, bs, nullptr, nullptr,
      nullptr, nullptr);
  gemm_wmma_bf16<1, 1, 0><<<dim3(NZP / 64, C / 128, B), blk, 0, stream>>>(
      Wg_bf, 0, C, zf_bf, sZF, NZP, ZG, sZF, NZP, NZ, C, bg, g_gamma, g_beta,
      g_mean, g_var);

  // ---- transpose xf_trans: [C][NXP] -> [NXP][C] ----
  transpose_bf16<<<dim3(NXP / 32, C / 32, B), blk, 0, stream>>>(XTcn, XTnc, C, NXP);

  // ---- attention logits ----
  gemm_wmma_bf16<0, 0, 1><<<dim3(NXP / 64, NXP / 128, B), blk, 0, stream>>>(
      XTnc, (long long)NXP * C, C, XTcn, sXF, NXP, Lself,
      (long long)NXP * NXP, NXP, NXP, C, nullptr, nullptr, nullptr, nullptr,
      nullptr);
  gemm_wmma_bf16<0, 0, 1><<<dim3(NZP / 64, NXP / 128, B), blk, 0, stream>>>(
      XTnc, (long long)NXP * C, C, ZT, sZF, NZP, Lsim, (long long)NXP * NZP,
      NZP, NZP, C, nullptr, nullptr, nullptr, nullptr, nullptr);

  // ---- softmax -> transposed bf16 probability matrices ----
  softmax_transpose<<<dim3(NXP, B), blk, 0, stream>>>(
      Lself, (long long)NXP * NXP, NXP, NX, NX, S_T, (long long)NXP * NXP, NXP,
      NXP);
  softmax_transpose<<<dim3(NXP, B), blk, 0, stream>>>(
      Lsim, (long long)NXP * NZP, NZP, NZ, NX, P_T, (long long)NZP * NXP, NXP,
      NZP);

  // ---- aggregation GEMMs produce [C,NX] directly into concat buffer F ----
  gemm_wmma_bf16<0, 0, 0><<<dim3(NXP / 64, C / 128, B), blk, 0, stream>>>(
      ZG, sZF, NZP, P_T, (long long)NZP * NXP, NXP, F, sF, NXP, NX, NZP,
      nullptr, nullptr, nullptr, nullptr, nullptr);
  gemm_wmma_bf16<0, 0, 0><<<dim3(NXP / 64, C / 128, B), blk, 0, stream>>>(
      F + (long long)512 * NXP, sF, NXP, S_T, (long long)NXP * NXP, NXP,
      F + (long long)256 * NXP, sF, NXP, NX, NXP, nullptr, nullptr, nullptr,
      nullptr, nullptr);

  // ---- final fused conv: out = relu(bn(Wfi * F + bfi)) , f32 to d_out ----
  gemm_wmma_bf16<1, 1, 1><<<dim3(NXP / 64, O / 128, B), blk, 0, stream>>>(
      Wfi_bf, 0, K3, F, sF, NXP, out, (long long)O * NX, NX, NX, K3, bfi,
      fi_gamma, fi_beta, fi_mean, fi_var);
}
